// simple_motionsqueeze_32117765439932
// MI455X (gfx1250) — matrix operand order repaired and checked by reading, not compiled, not run on hardware
//
#include <hip/hip_runtime.h>
#include <hip/hip_bf16.h>
#include <math.h>

typedef __attribute__((ext_vector_type(16))) _Float16 v16h;
typedef __attribute__((ext_vector_type(8)))  _Float16 v8h;
typedef __attribute__((ext_vector_type(8)))  float    v8f;

#define WMMA_F32_F16(a, b, c) \
  __builtin_amdgcn_wmma_f32_16x16x32_f16(false, (a), false, (b), (short)0, (c), false, false)

constexpr int N_ = 4, C_ = 256, T_ = 8, H_ = 28, W_ = 28, HW_ = 784;
constexpr int CP_ = 128, TP_ = 7;
constexpr int HO_ = 26, WO_ = 26, HWO_ = 676;
constexpr float INVNORM_ = 0.07978845608f;   // 1/(sqrt(2*pi)*5)
constexpr float INV_SIG2_ = 0.04f;           // 1/25
constexpr float INV_ETA_ = 100.0f;           // 1/0.01

// ---------------------------------------------------------------- utilities
__global__ void k_cvt_w(const float* __restrict__ w, _Float16* __restrict__ wq) {
  int i = blockIdx.x * 256 + threadIdx.x;
  if (i < CP_ * C_) wq[i] = (_Float16)w[i];
}

// conv weight transform: wt[tap][oc][ic] = w[oc][ic][tap]  (f32 OIHW -> f16 tap-major)
__global__ void k_cvt_conv_w(const float* __restrict__ w, _Float16* __restrict__ wt,
                             int Cout, int Cin) {
  int idx = blockIdx.x * 256 + threadIdx.x;
  int total = Cout * Cin * 9;
  if (idx >= total) return;
  int ic = idx % Cin; int t2 = idx / Cin;
  int oc = t2 % Cout; int kp = t2 / Cout;
  wt[idx] = (_Float16)w[((size_t)oc * Cin + ic) * 9 + kp];
}

__global__ void k_mean_y(const float* __restrict__ x, float* __restrict__ y) {
  __shared__ float s[256];
  int nc = blockIdx.x, tid = threadIdx.x;
  const float* p = x + (size_t)nc * (T_ * HW_);
  float acc = 0.f;
  for (int i = tid; i < T_ * HW_; i += 256) acc += p[i];
  s[tid] = acc; __syncthreads();
  for (int st = 128; st > 0; st >>= 1) { if (tid < st) s[tid] += s[tid + st]; __syncthreads(); }
  if (tid == 0) y[nc] = s[0] / (float)(T_ * HW_);
}

// ---------------------------------------------------------------- tm GEMM (WMMA)
__global__ void __launch_bounds__(256) k_tm_gemm(const float* __restrict__ x,
                                                 const _Float16* __restrict__ wq,
                                                 const float* __restrict__ cb,
                                                 _Float16* __restrict__ tm) {
  int wave = threadIdx.x >> 5;
  int lane = threadIdx.x & 31;
  int nt = blockIdx.x / 7;
  int tg = blockIdx.x % 7;
  int mt = tg * 8 + wave;          // pixel tile 0..55
  if (mt >= 49) return;            // wave-uniform exit
  int n = nt >> 3, t = nt & 7;
  int l15 = lane & 15;
  int hi  = (lane & 16) ? 1 : 0;
  int prow = mt * 16 + l15;
  const float* xr = x + (size_t)n * (C_ * T_ * HW_) + (size_t)t * HW_ + prow;

  v8f acc[8];
  for (int j = 0; j < 8; ++j)
    acc[j] = (v8f){0.f, 0.f, 0.f, 0.f, 0.f, 0.f, 0.f, 0.f};

  for (int k0 = 0; k0 < C_; k0 += 32) {
    union { v16h v; _Float16 e[16]; } a;
    int kb = k0 + hi * 16;
#pragma unroll
    for (int i = 0; i < 8; ++i) {
      float v0 = xr[(size_t)(kb + i) * (T_ * HW_)];
      float v1 = xr[(size_t)(kb + 8 + i) * (T_ * HW_)];
      a.e[i]     = (_Float16)fmaxf(v0, 0.f);
      a.e[8 + i] = (_Float16)fmaxf(v1, 0.f);
    }
    int kbB = k0 + hi * 16;
#pragma unroll
    for (int j = 0; j < 8; ++j) {
      int cp = j * 16 + l15;
      v16h b = *(const v16h*)(wq + (size_t)cp * C_ + kbB);
      acc[j] = WMMA_F32_F16(a.v, b, acc[j]);
    }
  }
  _Float16* out = tm + (size_t)nt * HW_ * CP_;
#pragma unroll
  for (int j = 0; j < 8; ++j) {
    int cp = j * 16 + l15;
    float bias = cb[cp];
#pragma unroll
    for (int r = 0; r < 8; ++r) {
      int pixel = mt * 16 + r + hi * 8;
      out[(size_t)pixel * CP_ + cp] = (_Float16)(acc[j][r] + bias);
    }
  }
}

// ---------------------------------------------------------------- fused score GEMM + flow
__global__ void __launch_bounds__(256) k_score_flow(const _Float16* __restrict__ tm,
                                                    float* __restrict__ sim) {
  __shared__ float srow[16][800];
  __shared__ float r1[256];
  __shared__ float r2[256];
  __shared__ int   ri[256];

  int qt = blockIdx.x % 49;
  int nt = blockIdx.x / 49;         // n*7 + t
  int n = nt / 7, t = nt % 7;
  int wave = threadIdx.x >> 5, lane = threadIdx.x & 31, tid = threadIdx.x;
  int l15 = lane & 15, hi = (lane & 16) ? 1 : 0;

  const _Float16* Abase = tm + (size_t)(n * T_ + t)     * HW_ * CP_;  // bf (rows q)
  const _Float16* Bbase = tm + (size_t)(n * T_ + t + 1) * HW_ * CP_;  // bh (rows p)
  const _Float16* arow = Abase + (size_t)(qt * 16 + l15) * CP_;

  for (int pt = wave; pt < 49; pt += 8) {
    v8f acc = (v8f){0.f, 0.f, 0.f, 0.f, 0.f, 0.f, 0.f, 0.f};
    const _Float16* brow = Bbase + (size_t)(pt * 16 + l15) * CP_;
#pragma unroll
    for (int k0 = 0; k0 < CP_; k0 += 32) {
      union { v16h v; v8h h[2]; } a;
      int kbA = k0 + hi * 16;
      a.h[0] = *(const v8h*)(arow + kbA);
      a.h[1] = *(const v8h*)(arow + kbA + 8);
      v16h b = *(const v16h*)(brow + k0 + hi * 16);
      acc = WMMA_F32_F16(a.v, b, acc);
    }
#pragma unroll
    for (int r = 0; r < 8; ++r)
      srow[r + hi * 8][pt * 16 + l15] = acc[r];
  }
  __syncthreads();

  for (int r = 0; r < 16; ++r) {
    int q = qt * 16 + r;
    float bm = -3.4e38f; int bi = 0;
    for (int i = tid; i < HW_; i += 256) {
      float v = srow[r][i];
      if (v > bm) { bm = v; bi = i; }
    }
    r1[tid] = bm; ri[tid] = bi; __syncthreads();
    for (int s = 128; s > 0; s >>= 1) {
      if (tid < s) {
        float b = r1[tid + s]; int ib = ri[tid + s];
        if (b > r1[tid] || (b == r1[tid] && ib < ri[tid])) { r1[tid] = b; ri[tid] = ib; }
      }
      __syncthreads();
    }
    float smax = r1[0]; int am = ri[0];
    __syncthreads();
    float ix = (float)(am / W_);
    float iy = (float)(am % W_);
    float m1 = -3.4e38f, m2 = -3.4e38f;
    for (int i = tid; i < HW_; i += 256) {
      float s_ = srow[r][i];
      float a = (float)(i / W_), b = (float)(i % W_);
      float gx = __expf((a - ix) * INV_SIG2_) * INVNORM_;
      float gy = __expf((b - iy) * INV_SIG2_) * INVNORM_;
      m1 = fmaxf(m1, gx * s_);
      m2 = fmaxf(m2, gy * s_);
    }
    r1[tid] = m1; r2[tid] = m2; __syncthreads();
    for (int s = 128; s > 0; s >>= 1) {
      if (tid < s) { r1[tid] = fmaxf(r1[tid], r1[tid + s]); r2[tid] = fmaxf(r2[tid], r2[tid + s]); }
      __syncthreads();
    }
    float M1 = r1[0], M2 = r2[0];
    __syncthreads();
    float ex = 0.f, nx = 0.f, ey = 0.f, ny = 0.f;
    for (int i = tid; i < HW_; i += 256) {
      float s_ = srow[r][i];
      float a = (float)(i / W_), b = (float)(i % W_);
      float gx = __expf((a - ix) * INV_SIG2_) * INVNORM_;
      float gy = __expf((b - iy) * INV_SIG2_) * INVNORM_;
      float e1 = __expf((gx * s_ - M1) * INV_ETA_);
      float e2 = __expf((gy * s_ - M2) * INV_ETA_);
      ex += e1; nx += a * e1;
      ey += e2; ny += b * e2;
    }
    r1[tid] = ex; r2[tid] = nx; __syncthreads();
    for (int s = 128; s > 0; s >>= 1) {
      if (tid < s) { r1[tid] += r1[tid + s]; r2[tid] += r2[tid + s]; }
      __syncthreads();
    }
    float EX = r1[0], NX = r2[0];
    __syncthreads();
    r1[tid] = ey; r2[tid] = ny; __syncthreads();
    for (int s = 128; s > 0; s >>= 1) {
      if (tid < s) { r1[tid] += r1[tid + s]; r2[tid] += r2[tid + s]; }
      __syncthreads();
    }
    if (tid == 0) {
      float EY = r1[0], NY = r2[0];
      float* sp = sim + (size_t)nt * 3 * HW_;
      sp[0 * HW_ + q] = fmaxf(NX / EX, 0.f);
      sp[1 * HW_ + q] = fmaxf(NY / EY, 0.f);
      sp[2 * HW_ + q] = fmaxf(smax, 0.f);
    }
    __syncthreads();
  }
}

// ---------------------------------------------------------------- conv1 (3->64, 5x5, pad 1), direct
// out f16 pixel-major [b][676][64]
__global__ void k_conv1(const float* __restrict__ sim, const float* __restrict__ w,
                        const float* __restrict__ bias, _Float16* __restrict__ out) {
  int idx = blockIdx.x * 256 + threadIdx.x;
  if (idx >= 28 * HWO_ * 64) return;
  int oc = idx & 63; int t2 = idx >> 6;
  int pixel = t2 % HWO_; int b = t2 / HWO_;
  int oy = pixel / WO_, ox = pixel % WO_;
  float acc = bias[oc];
  for (int ic = 0; ic < 3; ++ic) {
    const float* ip = sim + ((size_t)b * 3 + ic) * HW_;
    const float* wp = w + ((size_t)oc * 3 + ic) * 25;
    for (int ky = 0; ky < 5; ++ky) {
      int iy = oy + ky - 1;
      if (iy < 0 || iy >= H_) continue;
      for (int kx = 0; kx < 5; ++kx) {
        int ix = ox + kx - 1;
        if (ix < 0 || ix >= W_) continue;
        acc += ip[iy * W_ + ix] * wp[ky * 5 + kx];
      }
    }
  }
  out[idx] = (_Float16)fmaxf(acc, 0.f);
}

// ---------------------------------------------------------------- 3x3 implicit-GEMM conv (WMMA)
// act: f16 [b][676][Cin]; wt: f16 [9][Cout][Cin]; output pixel-major.
// Grid: (B, 6 m-blocks, Cout/128). Each wave: 16 pixels x 128 oc.
__global__ void __launch_bounds__(256) k_conv3x3(const _Float16* __restrict__ act,
                                                 const _Float16* __restrict__ wt,
                                                 const float* __restrict__ bias,
                                                 _Float16* __restrict__ outh,
                                                 float* __restrict__ outf,
                                                 int Cin, int Cout, int relu) {
  int wave = threadIdx.x >> 5, lane = threadIdx.x & 31;
  int l15 = lane & 15, hi = (lane & 16) ? 1 : 0;
  int b  = blockIdx.x;
  int mb = blockIdx.y;
  int ng = blockIdx.z;
  int mt = mb * 8 + wave;          // m tile 0..47
  if (mt >= 43) return;            // wave-uniform
  int m = mt * 16 + l15;           // output pixel for A-frag row
  int vm = (m < HWO_);
  int oy = m / WO_, ox = m % WO_;
  const _Float16* actb = act + (size_t)b * HWO_ * Cin;

  v8f acc[8];
#pragma unroll
  for (int j = 0; j < 8; ++j)
    acc[j] = (v8f){0.f, 0.f, 0.f, 0.f, 0.f, 0.f, 0.f, 0.f};

  for (int kp = 0; kp < 9; ++kp) {
    int ky = kp / 3 - 1, kx = kp % 3 - 1;
    int iy = oy + ky, ix = ox + kx;
    bool av = vm && iy >= 0 && iy < HO_ && ix >= 0 && ix < WO_;
    const _Float16* arow = actb + (size_t)(iy * WO_ + ix) * Cin;
    const _Float16* wrow = wt + (size_t)kp * Cout * Cin;
    for (int k0 = 0; k0 < Cin; k0 += 32) {
      union { v16h v; v8h h[2]; _Float16 e[16]; } a;
      if (av) {
        int kbA = k0 + hi * 16;
        a.h[0] = *(const v8h*)(arow + kbA);
        a.h[1] = *(const v8h*)(arow + kbA + 8);
      } else {
#pragma unroll
        for (int i = 0; i < 16; ++i) a.e[i] = (_Float16)0.f;
      }
      int kbB = k0 + hi * 16;
#pragma unroll
      for (int j = 0; j < 8; ++j) {
        int oc = ng * 128 + j * 16 + l15;
        v16h bf = *(const v16h*)(wrow + (size_t)oc * Cin + kbB);
        acc[j] = WMMA_F32_F16(a.v, bf, acc[j]);
      }
    }
  }
#pragma unroll
  for (int j = 0; j < 8; ++j) {
    int oc = ng * 128 + j * 16 + l15;
    float bv = bias[oc];
#pragma unroll
    for (int r = 0; r < 8; ++r) {
      int pixel = mt * 16 + r + hi * 8;
      if (pixel < HWO_) {
        float v = acc[j][r] + bv;
        if (relu) v = fmaxf(v, 0.f);
        size_t o = ((size_t)b * HWO_ + pixel) * Cout + oc;
        if (outh) outh[o] = (_Float16)v;
        else      outf[o] = v;
      }
    }
  }
}

// motion mean over pixel-major conv4 output [b][676][256]
__global__ void k_mean_motion(const float* __restrict__ c4, float* __restrict__ mv) {
  __shared__ float s[256];
  int nc = blockIdx.x, tid = threadIdx.x;
  int n = nc >> 8, c = nc & 255;
  float acc = 0.f;
  for (int t = 0; t < TP_; ++t) {
    const float* p = c4 + (size_t)(n * TP_ + t) * HWO_ * C_ + c;
    for (int i = tid; i < HWO_; i += 256) acc += p[(size_t)i * C_];
  }
  s[tid] = acc; __syncthreads();
  for (int st = 128; st > 0; st >>= 1) { if (tid < st) s[tid] += s[tid + st]; __syncthreads(); }
  if (tid == 0) mv[nc] = s[0] / (float)(TP_ * HWO_);
}

__global__ void k_final(const float* __restrict__ y, const float* __restrict__ mv,
                        const float* __restrict__ lw, const float* __restrict__ lb,
                        float* __restrict__ z) {
  __shared__ float a[256];
  int n = blockIdx.x, o = threadIdx.x;
  a[o] = fmaxf(y[n * 256 + o] + mv[n * 256 + o], 0.f);
  __syncthreads();
  float acc = lb[o];
  for (int c = 0; c < 256; ++c) acc += a[c] * lw[o * 256 + c];
  z[n * 256 + o] = acc;
}

// ---------------------------------------------------------------- launcher
extern "C" void kernel_launch(void* const* d_in, const int* in_sizes, int n_in,
                              void* d_out, int out_size, void* d_ws, size_t ws_size,
                              hipStream_t stream) {
  (void)in_sizes; (void)n_in; (void)out_size; (void)ws_size;
  const float* x      = (const float*)d_in[0];
  const float* corr_w = (const float*)d_in[1];
  const float* corr_b = (const float*)d_in[2];
  const float* mw1 = (const float*)d_in[3];  const float* mb1 = (const float*)d_in[4];
  const float* mw2 = (const float*)d_in[5];  const float* mb2 = (const float*)d_in[6];
  const float* mw3 = (const float*)d_in[7];  const float* mb3 = (const float*)d_in[8];
  const float* mw4 = (const float*)d_in[9];  const float* mb4 = (const float*)d_in[10];
  const float* lin_w = (const float*)d_in[11];
  const float* lin_b = (const float*)d_in[12];

  char* ws = (char*)d_ws;
  float*     y   = (float*)(ws + 0);                 //   4 KB
  float*     mv  = (float*)(ws + 4096);              //   4 KB
  _Float16*  wq  = (_Float16*)(ws + 8192);           //  64 KB corr_w f16
  _Float16*  tm  = (_Float16*)(ws + 73728);          // 6.1 MB [32][784][128] f16
  float*     sim = (float*)(ws + 6496256);           // 257 KB [28][3][784]
  _Float16*  a1  = (_Float16*)(ws + 6759680);        // 2.3 MB [28][676][64]
  _Float16*  a2  = (_Float16*)(ws + 9182464);        // 4.6 MB [28][676][128]
  _Float16*  a3  = (_Float16*)(ws + 14028032);       // 4.6 MB [28][676][128]
  float*     c4  = (float*)(ws + 18873600);          // 18.5 MB [28][676][256] f32
  _Float16*  wt2 = (_Float16*)(ws + 38255872);       // 144 KB [9][128][64]
  _Float16*  wt3 = (_Float16*)(ws + 38403328);       // 288 KB [9][128][128]
  _Float16*  wt4 = (_Float16*)(ws + 38698240);       // 576 KB [9][256][128]

  k_cvt_w<<<(CP_ * C_ + 255) / 256, 256, 0, stream>>>(corr_w, wq);
  k_cvt_conv_w<<<(128 * 64 * 9 + 255) / 256, 256, 0, stream>>>(mw2, wt2, 128, 64);
  k_cvt_conv_w<<<(128 * 128 * 9 + 255) / 256, 256, 0, stream>>>(mw3, wt3, 128, 128);
  k_cvt_conv_w<<<(256 * 128 * 9 + 255) / 256, 256, 0, stream>>>(mw4, wt4, 256, 128);
  k_mean_y<<<N_ * C_, 256, 0, stream>>>(x, y);
  k_tm_gemm<<<N_ * T_ * 7, 256, 0, stream>>>(x, wq, corr_b, tm);
  k_score_flow<<<N_ * TP_ * 49, 256, 0, stream>>>(tm, sim);
  k_conv1<<<(28 * HWO_ * 64 + 255) / 256, 256, 0, stream>>>(sim, mw1, mb1, a1);
  k_conv3x3<<<dim3(28, 6, 1), 256, 0, stream>>>(a1, wt2, mb2, a2, (float*)nullptr, 64, 128, 1);
  k_conv3x3<<<dim3(28, 6, 1), 256, 0, stream>>>(a2, wt3, mb3, a3, (float*)nullptr, 128, 128, 1);
  k_conv3x3<<<dim3(28, 6, 2), 256, 0, stream>>>(a3, wt4, mb4, (_Float16*)nullptr, c4, 128, 256, 0);
  k_mean_motion<<<N_ * C_, 256, 0, stream>>>(c4, mv);
  k_final<<<N_, 256, 0, stream>>>(y, mv, lin_w, lin_b, (float*)d_out);
}
